// Attention_3298534884255
// MI455X (gfx1250) — matrix operand order repaired and checked by reading, not compiled, not run on hardware
//
#include <hip/hip_runtime.h>
#include <hip/hip_bf16.h>

typedef __attribute__((ext_vector_type(16))) __bf16 v16bf;
typedef __attribute__((ext_vector_type(8)))  __bf16 v8bf;
typedef __attribute__((ext_vector_type(8)))  float  v8f;

#define S_LEN 2048
#define B_DIM 64
#define H_DIM 1024
#define NEG_VAL (-100000.0f)

// workspace layout (bytes)
#define AH_OFF   0                              // attn_hidden: B*H f32   = 256 KB
#define UBT_OFF  (B_DIM * H_DIM * 4)            // U^T bf16:    H*H bf16  = 2 MB
#define SC_OFF   (UBT_OFF + H_DIM * H_DIM * 2)  // scoresT:     B*S f32   = 512 KB

// hardware tanh (V_TANH_F32, TRANS op on gfx1250) with safe fallback
__device__ __forceinline__ float fast_tanh(float x) {
#if __has_builtin(__builtin_amdgcn_tanhf)
  return __builtin_amdgcn_tanhf(x);
#elif __has_builtin(__builtin_amdgcn_tanh_f32)
  return __builtin_amdgcn_tanh_f32(x);
#else
  return tanhf(x);
#endif
}

// ---------------------------------------------------------------------------
// Prep 1: attn_hidden[B,H] = hidden[0] @ W, bf16 WMMA, one wave per 16x16 tile
// ---------------------------------------------------------------------------
__global__ void __launch_bounds__(32)
prep_hidden_gemm(const float* __restrict__ hidden,
                 const float* __restrict__ W,
                 float* __restrict__ ah) {
  const int lane = threadIdx.x & 31;
  const int half = lane >> 4;    // 0: lanes 0-15, 1: lanes 16-31
  const int nl   = lane & 15;
  const int n0 = blockIdx.x * 16;
  const int m0 = blockIdx.y * 16;

  v8f acc = {0.f, 0.f, 0.f, 0.f, 0.f, 0.f, 0.f, 0.f};
  for (int k0 = 0; k0 < H_DIM; k0 += 32) {
    v16bf a, b;
    // A fragment: lane holds row m0+nl; K chunks {half*8..+7} and {half*8+16..+23}
    const int m = m0 + nl;
    #pragma unroll
    for (int i = 0; i < 8; ++i) {
      a[i]     = (__bf16)hidden[m * H_DIM + k0 + half * 8 + i];
      a[8 + i] = (__bf16)hidden[m * H_DIM + k0 + half * 8 + 16 + i];
    }
    // B fragment: lane holds column n0+nl; K chunks {half*8..+7} and {half*8+16..+23} (W is [K][N] row-major)
    const int n = n0 + nl;
    #pragma unroll
    for (int i = 0; i < 8; ++i) {
      b[i]     = (__bf16)W[(k0 + half * 8 + i) * H_DIM + n];
      b[8 + i] = (__bf16)W[(k0 + half * 8 + 16 + i) * H_DIM + n];
    }

    acc = __builtin_amdgcn_wmma_f32_16x16x32_bf16(false, a, false, b,
                                                  (short)0, acc, false, false);
  }
  // C layout: VGPR j -> M = j + 8*half, N = nl
  #pragma unroll
  for (int j = 0; j < 8; ++j)
    ah[(m0 + j + half * 8) * H_DIM + n0 + nl] = acc[j];
}

// ---------------------------------------------------------------------------
// Prep 2: Ubt[k*H + h] = bf16(U[h*H + k])   (so B tiles are K-contiguous)
// ---------------------------------------------------------------------------
__global__ void __launch_bounds__(256)
prep_transpose_u(const float* __restrict__ U, __bf16* __restrict__ ubt) {
  int idx = blockIdx.x * 256 + threadIdx.x;   // 0..H*H-1
  int k = idx >> 10;
  int h = idx & (H_DIM - 1);
  ubt[idx] = (__bf16)U[h * H_DIM + k];
}

// ---------------------------------------------------------------------------
// Fused: scoresT[b,s] = sum_n tanh( (inputs@U)[s,b,n] + ah[b,n] ) * v[n]
//   grid = S*B/128 WGs of 256 threads (8 waves); A tile 128x1024 bf16 in LDS
//   LDS = 256KB (A) + 32KB (B tile) + 4KB (v) + 4KB (ah slice) = 296KB
//   B tile staged via GLOBAL_LOAD_ASYNC_TO_LDS_B128 (ASYNCcnt path)
// ---------------------------------------------------------------------------
__global__ void __launch_bounds__(256)
fused_scores(const float* __restrict__ inputs,
             const float* __restrict__ ah,
             const __bf16* __restrict__ ubt,
             const float* __restrict__ v,
             float* __restrict__ scoresT) {
  extern __shared__ char smem[];
  __bf16* ldsA  = (__bf16*)smem;                                   // 128*1024 bf16
  __bf16* ldsB  = (__bf16*)(smem + 128 * H_DIM * 2);               // 16*1024 bf16
  float*  ldsV  = (float*)(smem + 128 * H_DIM * 2 + 16 * H_DIM * 2);       // 1024 f32
  float*  ldsAH = (float*)(smem + 128 * H_DIM * 2 + 16 * H_DIM * 2 + H_DIM * 4); // 64*16 f32

  const int tid  = threadIdx.x;
  const int wave = tid >> 5;
  const int lane = tid & 31;
  const int half = lane >> 4;
  const int nl   = lane & 15;
  const long long r0 = (long long)blockIdx.x * 128;   // first flat row (s*B+b)

  // Stage A tile: 128 rows x 1024 f32 -> bf16 LDS (float4 loads, b64 LDS stores)
  {
    const float4* src = (const float4*)(inputs + r0 * H_DIM);
    for (int i = tid; i < (128 * H_DIM) / 4; i += 256) {
      float4 f = src[i];
      __bf16* d = ldsA + i * 4;
      d[0] = (__bf16)f.x; d[1] = (__bf16)f.y;
      d[2] = (__bf16)f.z; d[3] = (__bf16)f.w;
    }
  }
  for (int i = tid; i < H_DIM; i += 256) ldsV[i] = v[i];

  // wave-relative LDS byte offset of ldsB (generic LDS ptr low 32 bits = LDS offset)
  const unsigned ldsB_off = (unsigned)(size_t)ldsB;

  float partial[8] = {0.f, 0.f, 0.f, 0.f, 0.f, 0.f, 0.f, 0.f};
  const int aRow = wave * 16 + nl;    // this lane's A row within the tile

  for (int nt = 0; nt < H_DIM / 16; ++nt) {
    const int n0 = nt * 16;
    __syncthreads();   // A ready (iter 0) / prior B+AH tiles consumed

    // Async-stage B tile: U^T rows n0..n0+15 (32KB bf16), 16B per async op,
    // 2048 chunks over 256 threads -> 8 GLOBAL_LOAD_ASYNC_TO_LDS_B128 per lane
    {
      const char* gB = (const char*)(ubt + (size_t)n0 * H_DIM);
      #pragma unroll
      for (int c = 0; c < 8; ++c) {
        const int i = tid + c * 256;            // chunk index, 16B chunks
        unsigned lds_addr = ldsB_off + i * 16;
        const char* gaddr = gB + i * 16;
        asm volatile("global_load_async_to_lds_b128 %0, %1, off"
                     :: "v"(lds_addr), "v"(gaddr) : "memory");
      }
    }
    // Stage attn_hidden slice for this n-tile: [64 b][16 n] f32 = 4KB
    for (int i = tid; i < B_DIM * 16; i += 256)
      ldsAH[i] = ah[(i >> 4) * H_DIM + n0 + (i & 15)];
    // Prefetch next U^T tile (one 128B line per thread -> global_prefetch_b8)
    if (nt + 1 < H_DIM / 16)
      __builtin_prefetch(ubt + (size_t)(n0 + 16) * H_DIM + tid * 64, 0, 1);

    asm volatile("s_wait_asynccnt 0" ::: "memory");
    __syncthreads();

    v8f acc = {0.f, 0.f, 0.f, 0.f, 0.f, 0.f, 0.f, 0.f};
    for (int k0 = 0; k0 < H_DIM; k0 += 32) {
      v16bf a, b;
      v8bf* ap = (v8bf*)&a;
      v8bf* bp = (v8bf*)&b;
      const __bf16* abase = ldsA + aRow * H_DIM + k0 + half * 8;
      ap[0] = *(const v8bf*)(abase);
      ap[1] = *(const v8bf*)(abase + 16);
      const __bf16* bbase = ldsB + nl * H_DIM + k0 + half * 8;
      bp[0] = *(const v8bf*)(bbase);
      bp[1] = *(const v8bf*)(bbase + 16);
      acc = __builtin_amdgcn_wmma_f32_16x16x32_bf16(false, a, false, b,
                                                    (short)0, acc, false, false);
    }

    // Epilogue: tanh(acc + ah[b,n]) * v[n]  (V_TANH_F32, LDS-resident operands)
    const float vn = ldsV[n0 + nl];
    #pragma unroll
    for (int j = 0; j < 8; ++j) {
      long long r = r0 + wave * 16 + j + half * 8;
      int b_idx = (int)(r & (B_DIM - 1));
      partial[j] += fast_tanh(acc[j] + ldsAH[b_idx * 16 + nl]) * vn;
    }
  }

  // Reduce over the 16 lanes of each half (xor masks stay within half)
  #pragma unroll
  for (int j = 0; j < 8; ++j) {
    float s = partial[j];
    s += __shfl_xor(s, 1);
    s += __shfl_xor(s, 2);
    s += __shfl_xor(s, 4);
    s += __shfl_xor(s, 8);
    partial[j] = s;
  }
  if (nl == 0) {
    #pragma unroll
    for (int j = 0; j < 8; ++j) {
      long long r = r0 + wave * 16 + j + half * 8;
      int b_idx = (int)(r & (B_DIM - 1));
      int s_idx = (int)(r >> 6);                      // r / B
      scoresT[(long long)b_idx * S_LEN + s_idx] = partial[j];
    }
  }
}

// ---------------------------------------------------------------------------
// Masked softmax over S per batch row
// ---------------------------------------------------------------------------
__global__ void __launch_bounds__(256)
softmax_kernel(const float* __restrict__ scoresT,
               const unsigned char* __restrict__ pad,
               float* __restrict__ out) {
  __shared__ float red[256];
  const int b = blockIdx.x;
  const int tid = threadIdx.x;

  float x[8];
  float mx = -3.4e38f;
  #pragma unroll
  for (int i = 0; i < 8; ++i) {
    int s = tid + i * 256;
    float val = pad[b * S_LEN + s] ? NEG_VAL : scoresT[b * S_LEN + s];
    x[i] = val;
    mx = fmaxf(mx, val);
  }
  red[tid] = mx;
  __syncthreads();
  for (int off = 128; off > 0; off >>= 1) {
    if (tid < off) red[tid] = fmaxf(red[tid], red[tid + off]);
    __syncthreads();
  }
  mx = red[0];
  __syncthreads();

  float sum = 0.f;
  #pragma unroll
  for (int i = 0; i < 8; ++i) { x[i] = __expf(x[i] - mx); sum += x[i]; }
  red[tid] = sum;
  __syncthreads();
  for (int off = 128; off > 0; off >>= 1) {
    if (tid < off) red[tid] += red[tid + off];
    __syncthreads();
  }
  const float inv = 1.0f / red[0];
  #pragma unroll
  for (int i = 0; i < 8; ++i)
    out[b * S_LEN + tid + i * 256] = x[i] * inv;
}

// ---------------------------------------------------------------------------
extern "C" void kernel_launch(void* const* d_in, const int* in_sizes, int n_in,
                              void* d_out, int out_size, void* d_ws, size_t ws_size,
                              hipStream_t stream) {
  const float* inputs         = (const float*)d_in[0];   // [S,B,H] f32
  const float* hidden         = (const float*)d_in[1];   // [1,B,H] f32
  const unsigned char* pad    = (const unsigned char*)d_in[2]; // [B,S] bool
  const float* W              = (const float*)d_in[3];   // [H,H] f32
  const float* U              = (const float*)d_in[4];   // [H,H] f32
  const float* v              = (const float*)d_in[5];   // [H,1] f32
  float* out                  = (float*)d_out;           // [B,S] f32

  char* ws = (char*)d_ws;
  float*  ah      = (float*)(ws + AH_OFF);
  __bf16* ubt     = (__bf16*)(ws + UBT_OFF);
  float*  scoresT = (float*)(ws + SC_OFF);

  // 1) attn_hidden = hidden @ W  (WMMA)
  prep_hidden_gemm<<<dim3(H_DIM / 16, B_DIM / 16), 32, 0, stream>>>(hidden, W, ah);
  // 2) U^T in bf16
  prep_transpose_u<<<(H_DIM * H_DIM) / 256, 256, 0, stream>>>(U, ubt);
  // 3) fused GEMM + tanh + dot(v) -> scoresT  (296KB dynamic LDS)
  const int ldsBytes = 128 * H_DIM * 2 + 16 * H_DIM * 2 + H_DIM * 4 + B_DIM * 16 * 4;
  hipFuncSetAttribute(reinterpret_cast<const void*>(fused_scores),
                      hipFuncAttributeMaxDynamicSharedMemorySize, ldsBytes);
  fused_scores<<<(S_LEN * B_DIM) / 128, 256, ldsBytes, stream>>>(inputs, ah, ubt, v, scoresT);
  // 4) masked softmax over S
  softmax_kernel<<<B_DIM, 256, 0, stream>>>(scoresT, pad, out);
}
